// MHA_3573412790784
// MI455X (gfx1250) — matrix operand order repaired and checked by reading, not compiled, not run on hardware
//
#include <hip/hip_runtime.h>
#include <hip/hip_bf16.h>

// MI455X / gfx1250 transformer block: bf16 WMMA GEMMs + block-cooperative
// flash attention. All matmuls lower to v_wmma_f32_16x16x32_bf16; softmax
// reductions use v_permlane16_b32; staging is double-buffered in LDS with
// one barrier per chunk and global_prefetch_b8 warming the next chunk.

typedef unsigned short u16;
typedef __attribute__((ext_vector_type(16))) __bf16 v16bf;
typedef __attribute__((ext_vector_type(8)))  float  v8f;
typedef __attribute__((ext_vector_type(4)))  float  v4f;
typedef __attribute__((ext_vector_type(16))) u16    v16u;
typedef __attribute__((ext_vector_type(8)))  u16    v8u;
typedef __attribute__((ext_vector_type(4)))  u16    v4u;

#define T_SEQ   4096
#define D_MODEL 768
#define NHEADS  12
#define HEAD_DIM 64
#define D_QKV   2304   // 3 * D_MODEL, layout [t][qkv*768 + head*64 + e]
#define D_FF    3072

#define CAT16(a, b) __builtin_shufflevector(a, b, 0,1,2,3,4,5,6,7,8,9,10,11,12,13,14,15)

__device__ __forceinline__ u16 f32_to_bf16(float f) {
  unsigned int u = __builtin_bit_cast(unsigned int, f);
  u += 0x7FFFu + ((u >> 16) & 1u);   // round-to-nearest-even
  return (u16)(u >> 16);
}

// XOR-butterfly within each 16-lane row: v_permlane16_b32 (pure VALU, can
// co-execute with the WMMA pipe) instead of ds_bpermute. Row-relative nibble
// tables: nibble_j = j ^ m.
#if defined(__has_builtin)
#if __has_builtin(__builtin_amdgcn_permlane16)
#define HAVE_PERMLANE16 1
#endif
#endif

template<int M>
__device__ __forceinline__ float bfly16(float v) {
#ifdef HAVE_PERMLANE16
  unsigned s0, s1;
  if      (M == 1) { s0 = 0x67452301u; s1 = 0xEFCDAB89u; }
  else if (M == 2) { s0 = 0x54761032u; s1 = 0xDCFE98BAu; }
  else if (M == 4) { s0 = 0x32107654u; s1 = 0xBA98FEDCu; }
  else             { s0 = 0xFEDCBA98u; s1 = 0x76543210u; }
  int i = __builtin_bit_cast(int, v);
  i = __builtin_amdgcn_permlane16(i, i, s0, s1, false, false);
  return __builtin_bit_cast(float, i);
#else
  return __shfl_xor(v, M, 32);
#endif
}

__device__ __forceinline__ float half16_max(float v) {
  v = fmaxf(v, bfly16<1>(v));
  v = fmaxf(v, bfly16<2>(v));
  v = fmaxf(v, bfly16<4>(v));
  v = fmaxf(v, bfly16<8>(v));
  return v;
}
__device__ __forceinline__ float half16_sum(float v) {
  v += bfly16<1>(v);
  v += bfly16<2>(v);
  v += bfly16<4>(v);
  v += bfly16<8>(v);
  return v;
}

__global__ void cast_f32_bf16(const float* __restrict__ in, u16* __restrict__ out, int n4) {
  int i = blockIdx.x * 256 + threadIdx.x;
  if (i < n4) {
    v4f f = ((const v4f*)in)[i];
    v4u o;
    #pragma unroll
    for (int j = 0; j < 4; j++) o[j] = f32_to_bf16(f[j]);
    ((v4u*)out)[i] = o;
  }
}

// C = A(bf16, MxK rm) @ B(bf16, KxN rm) + bias, optional exact GELU,
// bf16 or f32 out, ldc == N. 256 thr = 8 waves; tile 128x128; BK=32.
// A fragments load straight from global (no cross-wave reuse); B is staged
// transposed in double-buffered LDS -> one barrier per K-step.
template<bool GELU, bool OUTF32>
__global__ __launch_bounds__(256) void gemm_bf16_wmma(
    const u16* __restrict__ A, const u16* __restrict__ B,
    const float* __restrict__ bias, void* __restrict__ Cout,
    int M, int N, int K)
{
  __shared__ __align__(64) u16 Bt[2][128 * 32];   // [n][k] transposed, 2 x 8 KB

  const int tid  = threadIdx.x;
  const int wave = tid >> 5;
  const int lane = tid & 31;
  const int hl   = lane & 15;
  const bool lo  = lane < 16;
  const int bm = blockIdx.y * 128;
  const int bn = blockIdx.x * 128;

  const int kk = tid >> 3;            // B staging: k row 0..31
  const int n0 = (tid & 7) << 4;      // B staging: n col 0,16,...,112
  const u16* brow = B + (size_t)kk * N + bn + n0;           // this thread's B src
  const u16* arow = A + (size_t)(bm + wave * 16 + hl) * K;  // this lane's A row
  const int ao = lo ? 0 : 16;         // A-frag K sub-offset per lane half

  v8f acc[8] = {};
  int buf = 0;
  for (int k0 = 0; k0 < K; k0 += 32, buf ^= 1) {
    { // stage B transposed: coalesced b128 row reads, 16-bit LDS scatter
      const u16* src = brow + (size_t)k0 * N;
      v8u b0 = *(const v8u*)src;
      v8u b1 = *(const v8u*)(src + 8);
      #pragma unroll
      for (int i = 0; i < 8; i++) Bt[buf][(n0 + i) * 32 + kk]     = b0[i];
      #pragma unroll
      for (int i = 0; i < 8; i++) Bt[buf][(n0 + 8 + i) * 32 + kk] = b1[i];
    }
    if (k0 + 32 < K) {   // warm next chunk (global_prefetch_b8, uniform cond)
      __builtin_prefetch(brow + (size_t)(k0 + 32) * N, 0, 3);
      __builtin_prefetch(arow + k0 + 32, 0, 3);
    }
    __syncthreads();   // single barrier; double buffer covers WAR across iters

    // A fragment 16x32: lanes 0-15 hold K=0..15, lanes 16-31 K=16..31
    v8u x0 = *(const v8u*)(arow + k0 + ao);
    v8u x1 = *(const v8u*)(arow + k0 + 8 + ao);
    v16u af = CAT16(x0, x1);
    v16bf a = __builtin_bit_cast(v16bf, af);

    #pragma unroll
    for (int nt = 0; nt < 8; nt++) {
      // B fragment 32x16: lane col = hl; lanes 0-15 K=0..15, 16-31 K=16..31
      v16u bu = *(const v16u*)&Bt[buf][(nt * 16 + hl) * 32 + (lo ? 0 : 16)];
      acc[nt] = __builtin_amdgcn_wmma_f32_16x16x32_bf16(false, a, false,
                  __builtin_bit_cast(v16bf, bu), (short)0, acc[nt], false, false);
    }
  }

  // C layout: VGPR r -> row r (lanes 0-15) / r+8 (lanes 16-31), col = hl
  #pragma unroll
  for (int nt = 0; nt < 8; nt++) {
    int col = bn + nt * 16 + hl;
    float bv = bias[col];
    #pragma unroll
    for (int r = 0; r < 8; r++) {
      int row = bm + wave * 16 + r + (lo ? 0 : 8);
      float v = acc[nt][r] + bv;
      if (GELU) v = 0.5f * v * (1.0f + erff(v * 0.70710678118654752f));
      if (OUTF32) ((float*)Cout)[(size_t)row * N + col] = v;
      else        ((u16*)Cout)[(size_t)row * N + col]   = f32_to_bf16(v);
    }
  }
}

// Flash attention. Block = 8 waves, all on the SAME head (32 blocks/head),
// each wave owns a 16-row query tile; all waves step key chunks in lockstep,
// so K/V chunks are staged in LDS once per block (double-buffered, one
// barrier per chunk). V staged transposed -> contiguous B-fragment ds reads.
__global__ __launch_bounds__(256) void attn_wmma(const u16* __restrict__ qkv,
                                                 u16* __restrict__ attnOut)
{
  __shared__ __align__(64) u16 Ks[2][32 * 64];   // [key][e]   2 x 4 KB
  __shared__ __align__(64) u16 Vt[2][64 * 32];   // [e][key]   2 x 4 KB
  __shared__ __align__(64) u16 P[8][16 * 32];    // per-wave P relayout, 8 KB

  const int tid  = threadIdx.x;
  const int lane = tid & 31;
  const int wave = tid >> 5;
  const int hl   = lane & 15;
  const bool lo  = lane < 16;
  const int head = blockIdx.x >> 5;
  const int row0 = ((blockIdx.x & 31) * 8 + wave) * 16;
  const float scale = 0.03608439182435161f;      // 1/sqrt(768) per reference

  const int srow = tid >> 3;                     // staging: key 0..31
  const int scol = (tid & 7) << 3;               // staging: e 0,8,...,56
  const u16* kstage = qkv + D_MODEL     + head * HEAD_DIM + scol;
  const u16* vstage = qkv + 2 * D_MODEL + head * HEAD_DIM + scol;

  // Q fragments: two 16x32 A-frags covering hd=64, loaded once
  v16bf qf[2];
  {
    const u16* qr = qkv + (size_t)(row0 + hl) * D_QKV + head * HEAD_DIM;
    #pragma unroll
    for (int c = 0; c < 2; c++) {
      int o = c * 32 + (lo ? 0 : 16);
      v8u x0 = *(const v8u*)(qr + o);
      v8u x1 = *(const v8u*)(qr + o + 8);
      v16u t = CAT16(x0, x1);
      qf[c] = __builtin_bit_cast(v16bf, t);
    }
  }

  float mrow[8], lrow[8];
  #pragma unroll
  for (int r = 0; r < 8; r++) { mrow[r] = -3.0e38f; lrow[r] = 0.0f; }
  v8f Oacc[4] = {};

  int buf = 0;
  for (int kc = 0; kc < T_SEQ / 32; kc++, buf ^= 1) {
    { // cooperative K/V stage: coalesced b128 reads; V scattered transposed
      const size_t roff = (size_t)(kc * 32 + srow) * D_QKV;
      v8u kv = *(const v8u*)(kstage + roff);
      v8u vv = *(const v8u*)(vstage + roff);
      *(v8u*)&Ks[buf][srow * 64 + scol] = kv;
      #pragma unroll
      for (int i = 0; i < 8; i++) Vt[buf][(scol + i) * 32 + srow] = vv[i];
    }
    if (kc + 1 < T_SEQ / 32) {  // warm next K/V chunk (uniform condition)
      const size_t noff = (size_t)((kc + 1) * 32 + srow) * D_QKV;
      __builtin_prefetch(kstage + noff, 0, 3);
      __builtin_prefetch(vstage + noff, 0, 3);
    }
    __syncthreads();

    // S = Q K^T : two 16x16 score tiles, contraction hd=64 -> 2 WMMAs each
    v8f s[2] = {};
    #pragma unroll
    for (int st = 0; st < 2; st++) {
      #pragma unroll
      for (int c = 0; c < 2; c++) {
        v16u kb = *(const v16u*)&Ks[buf][(st * 16 + hl) * 64 + c * 32 + (lo ? 0 : 16)];
        s[st] = __builtin_amdgcn_wmma_f32_16x16x32_bf16(false, qf[c], false,
                  __builtin_bit_cast(v16bf, kb), (short)0, s[st], false, false);
      }
    }

    // Online softmax: rows live in (reg r, lane-half); permlane16 butterflies
    #pragma unroll
    for (int r = 0; r < 8; r++) {
      float s0 = s[0][r] * scale;
      float s1 = s[1][r] * scale;
      float mx   = half16_max(fmaxf(s0, s1));
      float mnew = fmaxf(mrow[r], mx);
      float alpha = __expf(mrow[r] - mnew);
      float p0 = __expf(s0 - mnew);
      float p1 = __expf(s1 - mnew);
      lrow[r] = lrow[r] * alpha + half16_sum(p0 + p1);
      mrow[r] = mnew;
      #pragma unroll
      for (int nt = 0; nt < 4; nt++) Oacc[nt][r] *= alpha;
      int prow = r + (lo ? 0 : 8);
      P[wave][prow * 32 + hl]      = f32_to_bf16(p0);
      P[wave][prow * 32 + 16 + hl] = f32_to_bf16(p1);
    }

    // P as A-fragment: own-wave LDS region, same-wave DS ops are in-order ->
    // no barrier needed between the stores above and these loads.
    v16bf pa;
    {
      int o = lo ? 0 : 16;
      v8u x0 = *(const v8u*)&P[wave][hl * 32 + o];
      v8u x1 = *(const v8u*)&P[wave][hl * 32 + 8 + o];
      v16u t = CAT16(x0, x1);
      pa = __builtin_bit_cast(v16bf, t);
    }

    // O += P @ V : contiguous B-frags from transposed V in LDS
    #pragma unroll
    for (int nt = 0; nt < 4; nt++) {
      v16u vb = *(const v16u*)&Vt[buf][(nt * 16 + hl) * 32 + (lo ? 0 : 16)];
      Oacc[nt] = __builtin_amdgcn_wmma_f32_16x16x32_bf16(false, pa, false,
                   __builtin_bit_cast(v16bf, vb), (short)0, Oacc[nt], false, false);
    }
  }

  #pragma unroll
  for (int nt = 0; nt < 4; nt++) {
    #pragma unroll
    for (int r = 0; r < 8; r++) {
      int row = row0 + r + (lo ? 0 : 8);
      float o = Oacc[nt][r] / lrow[r];
      attnOut[(size_t)row * D_MODEL + head * HEAD_DIM + nt * 16 + hl] = f32_to_bf16(o);
    }
  }
}

extern "C" void kernel_launch(void* const* d_in, const int* in_sizes, int n_in,
                              void* d_out, int out_size, void* d_ws, size_t ws_size,
                              hipStream_t stream)
{
  (void)in_sizes; (void)n_in; (void)out_size; (void)ws_size;
  const float* x     = (const float*)d_in[0];
  const float* w_qkv = (const float*)d_in[1];
  const float* b_qkv = (const float*)d_in[2];
  const float* w_ff1 = (const float*)d_in[3];
  const float* b_ff1 = (const float*)d_in[4];
  const float* w_ff2 = (const float*)d_in[5];
  const float* b_ff2 = (const float*)d_in[6];
  float* out = (float*)d_out;

  char* ws = (char*)d_ws;
  size_t off = 0;
  auto carve = [&](size_t elems) -> u16* {
    u16* p = (u16*)(ws + off);
    off += (elems * sizeof(u16) + 255) & ~(size_t)255;
    return p;
  };
  u16* xb    = carve((size_t)T_SEQ * D_MODEL);
  u16* wqkvb = carve((size_t)D_MODEL * D_QKV);
  u16* wff1b = carve((size_t)D_MODEL * D_FF);
  u16* wff2b = carve((size_t)D_FF * D_MODEL);
  u16* qkvb  = carve((size_t)T_SEQ * D_QKV);
  u16* attnb = carve((size_t)T_SEQ * D_MODEL);
  u16* hb    = carve((size_t)T_SEQ * D_FF);
  // ~70 MB total -- fully L2-resident on MI455X (192 MB)

  auto castN = [&](const float* src, u16* dst, int n) {
    cast_f32_bf16<<<(n / 4 + 255) / 256, 256, 0, stream>>>(src, dst, n / 4);
  };
  castN(x,     xb,    T_SEQ * D_MODEL);
  castN(w_qkv, wqkvb, D_MODEL * D_QKV);
  castN(w_ff1, wff1b, D_MODEL * D_FF);
  castN(w_ff2, wff2b, D_FF * D_MODEL);

  // qkv = x @ w_qkv + b_qkv                      (4096x2304, K=768)
  gemm_bf16_wmma<false, false><<<dim3(D_QKV / 128, T_SEQ / 128), 256, 0, stream>>>(
      xb, wqkvb, b_qkv, qkvb, T_SEQ, D_QKV, D_MODEL);

  // flash attention: 12 heads x 32 blocks, 8 query tiles per block
  attn_wmma<<<NHEADS * 32, 256, 0, stream>>>(qkvb, attnb);

  // h = gelu(attn @ w_ff1 + b_ff1)               (4096x3072, K=768)
  gemm_bf16_wmma<true, false><<<dim3(D_FF / 128, T_SEQ / 128), 256, 0, stream>>>(
      attnb, wff1b, b_ff1, hb, T_SEQ, D_FF, D_MODEL);

  // out = h @ w_ff2 + b_ff2  (f32)               (4096x768, K=3072)
  gemm_bf16_wmma<false, true><<<dim3(D_MODEL / 128, T_SEQ / 128), 256, 0, stream>>>(
      hb, wff2b, b_ff2, out, T_SEQ, D_MODEL, D_FF);
}
